// TrainingScriptAttention_70686571758175
// MI455X (gfx1250) — matrix operand order repaired and checked by reading, not compiled, not run on hardware
//
#include <hip/hip_runtime.h>

// ---------------------------------------------------------------------------
// Types
// ---------------------------------------------------------------------------
typedef __bf16 bf16;
typedef __attribute__((ext_vector_type(16))) __bf16 v16bf;
typedef __attribute__((ext_vector_type(8)))  __bf16 v8bf;
typedef __attribute__((ext_vector_type(8)))  float  v8f;
typedef __attribute__((ext_vector_type(4)))  unsigned int v4u;
typedef __attribute__((ext_vector_type(4)))  int v4i;
typedef __attribute__((ext_vector_type(8)))  int v8i;

#define B_SZ 2
#define T_SZ 4096
#define D_SZ 1024
#define H_SZ 8
#define HD_SZ 128

#if defined(__has_builtin)
#  if __has_builtin(__builtin_amdgcn_tensor_load_to_lds)
#    define HAVE_TDM 1
#  else
#    define HAVE_TDM 0
#  endif
#else
#  define HAVE_TDM 0
#endif

#if defined(__HIP_DEVICE_COMPILE__) && !HAVE_TDM
#warning "CDNA5 probe: __builtin_amdgcn_tensor_load_to_lds NOT available, using fallback K-tile copy"
#endif

// ---------------------------------------------------------------------------
// Fragment helpers (layouts per CDNA5 ISA 7.12.2)
// A 16x32 bf16 : lanes 0-15 -> M=lane, K = 0..7 & 16..23
//               lanes16-31 -> M=lane-16, K = 8..15 & 24..31
// B 32x16 bf16 : lanes 0-15 -> N=lane, K = 0..7 & 16..23 ; lanes 16-31 -> K = 8..15 & 24..31
// C/D 16x16 f32: VGPR r, lanes 0-15 -> M=r, N=lane ; lanes 16-31 -> M=8+r
// ---------------------------------------------------------------------------
__device__ __forceinline__ v16bf load_frag(const bf16* p0, const bf16* p1) {
  v16bf r;
  *(v8bf*)&r       = *(const v8bf*)p0;
  *((v8bf*)&r + 1) = *(const v8bf*)p1;
  return r;
}

// A-frag: row fixed per lane (pass row base pointer), K chunk by lane group
__device__ __forceinline__ v16bf a_frag(const bf16* rowp, int lane) {
  const int k0 = (lane >> 4) * 8;
  return load_frag(rowp + k0, rowp + k0 + 16);
}

// B-frag: col-major-by-lane storage [n][k]; pass full per-lane column index
__device__ __forceinline__ v16bf b_frag(const bf16* base, int col, int stride, int lane) {
  const bf16* p = base + col * stride + (lane >> 4) * 8;
  return load_frag(p, p + 16);
}

__device__ __forceinline__ v8f wmma_bf16(v16bf a, v16bf b, v8f c) {
  return __builtin_amdgcn_wmma_f32_16x16x32_bf16(false, a, false, b,
                                                 (short)0, c, false, false);
}

// Stage a 128x32 fp32 tile -> bf16 LDS tile with stride 40 (16B-aligned rows)
__device__ __forceinline__ void stage_f32_bf16(bf16* lds, const float* g, int tid) {
  const int row  = tid >> 1;
  const int col0 = (tid & 1) * 16;
  const float4* q4 = (const float4*)(g + (long long)row * 1024 + col0);
  float fv[16];
  ((float4*)fv)[0] = q4[0];
  ((float4*)fv)[1] = q4[1];
  ((float4*)fv)[2] = q4[2];
  ((float4*)fv)[3] = q4[3];
  v8bf lo, hi;
#pragma unroll
  for (int j = 0; j < 8; ++j) { lo[j] = (bf16)fv[j]; hi[j] = (bf16)fv[8 + j]; }
  v8bf* d = (v8bf*)&lds[row * 40 + col0];
  d[0] = lo; d[1] = hi;
}

// ---------------------------------------------------------------------------
// Kernel 1: QKV projection + fused RoPE.  M=8192 (b*t), N=3072 (c*h*d), K=1024
// Block 256 thr = 8 waves (2 M x 4 N). Wave owns 64 rows x cols {16w..16w+15}
// and {64+16w..64+16w+15} so RoPE pairs (d, d+64) stay in-wave.
// ---------------------------------------------------------------------------
__global__ __launch_bounds__(256) void qkv_rope_kernel(
    const float* __restrict__ x, const float* __restrict__ w,
    bf16* __restrict__ qo, bf16* __restrict__ ko, bf16* __restrict__ vo) {
  __shared__ bf16 aT[128 * 40];
  __shared__ bf16 bT[128 * 40];
  const int tid = threadIdx.x, lane = tid & 31, wv = tid >> 5;
  const int waveM = wv >> 2, waveN = wv & 3;
  const int mT = blockIdx.x, nT = blockIdx.y;      // 64 x 24
  const int c = nT >> 3, h = nT & 7;

  const float* xa = x + (long long)(mT * 128) * 1024;
  const float* wa = w + (long long)(nT * 128) * 1024;

  v8f acc[4][2] = {};
  for (int kt = 0; kt < 1024; kt += 32) {
    stage_f32_bf16(aT, xa + kt, tid);
    stage_f32_bf16(bT, wa + kt, tid);
    if (kt + 32 < 1024) {  // prefetch next K tile (global_prefetch_b8)
      __builtin_prefetch(xa + (long long)(tid >> 1) * 1024 + kt + 32 + (tid & 1) * 16, 0, 1);
      __builtin_prefetch(wa + (long long)(tid >> 1) * 1024 + kt + 32 + (tid & 1) * 16, 0, 1);
    }
    __syncthreads();
    const v16bf b0 = b_frag(bT, waveN * 16 + (lane & 15), 40, lane);
    const v16bf b1 = b_frag(bT, 64 + waveN * 16 + (lane & 15), 40, lane);
#pragma unroll
    for (int mf = 0; mf < 4; ++mf) {
      const v16bf a = a_frag(&aT[(waveM * 64 + mf * 16 + (lane & 15)) * 40], lane);
      acc[mf][0] = wmma_bf16(a, b0, acc[mf][0]);
      acc[mf][1] = wmma_bf16(a, b1, acc[mf][1]);
    }
    __syncthreads();
  }

  // Epilogue: RoPE (q,k) or passthrough (v), store bf16 [b,h,t,d]
  bf16* ob = (c == 0) ? qo : ((c == 1) ? ko : vo);
  const int d0 = waveN * 16 + (lane & 15);                 // 0..63
  const float fr = (d0 < 32) ? __expf(-0.22359586469675656f * (float)d0) : 0.0f;
#pragma unroll
  for (int mf = 0; mf < 4; ++mf) {
#pragma unroll
    for (int r = 0; r < 8; ++r) {
      const int rowM = mf * 16 + (lane >> 4) * 8 + r;
      const int grow = mT * 128 + waveM * 64 + rowM;
      const int bb = grow >> 12, t = grow & 4095;
      const float x1 = acc[mf][0][r], x2 = acc[mf][1][r];
      float y1, y2;
      if (c < 2) {
        const float th = fr * (float)t;
        const float cs = __cosf(th), sn = __sinf(th);
        y1 = x1 * cs + x2 * sn;
        y2 = -x1 * sn + x2 * cs;
      } else { y1 = x1; y2 = x2; }
      const long long base = (((long long)(bb * H_SZ + h)) * T_SZ + t) * HD_SZ;
      ob[base + d0]      = (bf16)y1;
      ob[base + 64 + d0] = (bf16)y2;
    }
  }
}

// ---------------------------------------------------------------------------
// Kernel 2: causal flash attention, one (b,h) per blockIdx.y, 128 rows per
// block (16 rows / wave). Key tiles of 32: K staged via TDM, V transposed.
// ---------------------------------------------------------------------------
#define KLS 136   // 128 + 8 pad (16B), matches TDM pad_interval=64dw/pad=4dw
#define VTS 40
#define PLS 40

__global__ __launch_bounds__(256) void flash_attn_kernel(
    const bf16* __restrict__ qg, const bf16* __restrict__ kg,
    const bf16* __restrict__ vg, bf16* __restrict__ og) {
  __shared__ bf16 kl[32 * KLS];
  __shared__ bf16 vt[128 * VTS];
  __shared__ bf16 pl[128 * PLS];
  const int tid = threadIdx.x, lane = tid & 31, wv = tid >> 5;
  const int bh = blockIdx.y;
  const long long hb = (long long)bh * T_SZ * HD_SZ;
  const int m0 = blockIdx.x * 128 + wv * 16;
  const int rbase = (lane >> 4) * 8;

  // per-wave Q fragments (16 rows x 128 d -> 4 A-frags)
  v16bf qa[4];
  {
    const bf16* qrow = qg + hb + (long long)(m0 + (lane & 15)) * HD_SZ;
#pragma unroll
    for (int kf = 0; kf < 4; ++kf) qa[kf] = a_frag(qrow + kf * 32, lane);
  }

  v8f o[8] = {};
  float mi[8], li[8];
#pragma unroll
  for (int r = 0; r < 8; ++r) { mi[r] = -__builtin_inff(); li[r] = 0.0f; }

  const int nkt = (blockIdx.x + 1) * 4;            // causal: keys <= block max row
  for (int ktile = 0; ktile < nkt; ++ktile) {
    const int s0 = ktile * 32;
    { // stage V transposed: vt[d][key]
      const int row = tid >> 3, col0 = (tid & 7) * 16;
      const bf16* gp = vg + hb + (long long)(s0 + row) * HD_SZ + col0;
      const v8bf u0 = *(const v8bf*)gp;
      const v8bf u1 = *(const v8bf*)(gp + 8);
#pragma unroll
      for (int j = 0; j < 8; ++j) {
        vt[(col0 + j) * VTS + row]     = u0[j];
        vt[(col0 + 8 + j) * VTS + row] = u1[j];
      }
      if (ktile + 1 < nkt) {  // prefetch next key tile of K and V
        const long long nxt = hb + (long long)(s0 + 32 + row) * HD_SZ + col0;
        __builtin_prefetch(vg + nxt, 0, 1);
        __builtin_prefetch(kg + nxt, 0, 1);
      }
    }
#if HAVE_TDM
    if (wv == 0) { // Tensor Data Mover: 32x128 bf16 tile -> LDS with row pad
      const unsigned lds_off = (unsigned)(unsigned long long)(&kl[0]);
      const unsigned long long ga =
          (unsigned long long)(kg + hb + (long long)s0 * HD_SZ);
      v4u g0;
      g0[0] = 1u;                                   // count=1
      g0[1] = lds_off;
      g0[2] = (unsigned)ga;
      g0[3] = (unsigned)(ga >> 32) | (2u << 30);    // type=2 (image)
      v8i g1;
      g1[0] = (int)((1u << 16) |                    // data_size = 2B
                    (1u << 20) |                    // pad_enable
                    (5u << 22) |                    // pad_interval = 64 dwords
                    (3u << 25));                    // pad_amount  = 4 dwords
      g1[1] = (int)(128u << 16);                    // tensor_dim0 = 128
      g1[2] = (int)(4096u << 16);                   // tensor_dim1 = 4096
      g1[3] = (int)(128u << 16);                    // tile_dim0   = 128
      g1[4] = (int)32u;                             // tile_dim1   = 32
      g1[5] = (int)128u;                            // tensor_dim0_stride = 128
      g1[6] = 0; g1[7] = 0;
      const v4i gz = {0, 0, 0, 0};
#if __clang_major__ >= 23
      const v8i gz8 = {0, 0, 0, 0, 0, 0, 0, 0};
      __builtin_amdgcn_tensor_load_to_lds(g0, g1, gz, gz, gz8, 0);
#else
      __builtin_amdgcn_tensor_load_to_lds(g0, g1, gz, gz, 0);
#endif
      __builtin_amdgcn_s_wait_tensorcnt(0);
    }
#else
    { // fallback cooperative copy of K tile
      const int row = tid >> 3, col0 = (tid & 7) * 16;
      const bf16* gp = kg + hb + (long long)(s0 + row) * HD_SZ + col0;
      v8bf* d = (v8bf*)&kl[row * KLS + col0];
      d[0] = *(const v8bf*)gp;
      d[1] = *(const v8bf*)(gp + 8);
    }
#endif
    __syncthreads();

    // scores: S = Q K^T (two 16x16 key frags, K-dim = head_dim in 4 steps)
    v8f sc0 = {}, sc1 = {};
#pragma unroll
    for (int kf = 0; kf < 4; ++kf) {
      const v16bf b0 = b_frag(kl + kf * 32, (lane & 15), KLS, lane);
      const v16bf b1 = b_frag(kl + kf * 32, 16 + (lane & 15), KLS, lane);
      sc0 = wmma_bf16(qa[kf], b0, sc0);
      sc1 = wmma_bf16(qa[kf], b1, sc1);
    }

    // online softmax (row = rbase+r, 16 lanes per row)
    const float scale = 0.08838834764831845f;      // 1/sqrt(128)
#pragma unroll
    for (int r = 0; r < 8; ++r) {
      const int t = m0 + rbase + r;
      float v0 = sc0[r] * scale, v1 = sc1[r] * scale;
      if (s0 + (lane & 15) > t)      v0 = -__builtin_inff();
      if (s0 + 16 + (lane & 15) > t) v1 = -__builtin_inff();
      float mx = fmaxf(v0, v1);
#pragma unroll
      for (int off = 8; off; off >>= 1) mx = fmaxf(mx, __shfl_xor(mx, off, 32));
      const float mnew  = fmaxf(mi[r], mx);
      const float alpha = __expf(mi[r] - mnew);
      const float p0 = __expf(v0 - mnew);
      const float p1 = __expf(v1 - mnew);
      float rs = p0 + p1;
#pragma unroll
      for (int off = 8; off; off >>= 1) rs += __shfl_xor(rs, off, 32);
      li[r] = li[r] * alpha + rs;
      mi[r] = mnew;
#pragma unroll
      for (int f = 0; f < 8; ++f) o[f][r] = o[f][r] * alpha;
      bf16* pr = &pl[(wv * 16 + rbase + r) * PLS];
      pr[lane & 15]        = (bf16)p0;
      pr[16 + (lane & 15)] = (bf16)p1;
    }
    asm volatile("s_wait_dscnt 0x0" ::: "memory");  // per-wave LDS RAW fence

    // P (A-frag) x V (B-frags) -> O accumulators
    const v16bf pa = a_frag(&pl[(wv * 16 + (lane & 15)) * PLS], lane);
#pragma unroll
    for (int f = 0; f < 8; ++f) {
      const v16bf b = b_frag(vt, f * 16 + (lane & 15), VTS, lane);
      o[f] = wmma_bf16(pa, b, o[f]);
    }
    __syncthreads();
  }

  // epilogue: O / l, store bf16 [b, t, h*128+d]
  const int bb = bh >> 3, h = bh & 7;
#pragma unroll
  for (int r = 0; r < 8; ++r) {
    const float inv = 1.0f / li[r];
    const int t = m0 + rbase + r;
    const long long base = ((long long)bb * T_SZ + t) * (long long)D_SZ + h * HD_SZ;
#pragma unroll
    for (int f = 0; f < 8; ++f)
      og[base + f * 16 + (lane & 15)] = (bf16)(o[f][r] * inv);
  }
}

// ---------------------------------------------------------------------------
// Kernel 3: output projection.  M=8192, N=1024, K=1024, fp32 out.
// ---------------------------------------------------------------------------
__global__ __launch_bounds__(256) void proj_kernel(
    const bf16* __restrict__ ag, const float* __restrict__ w,
    float* __restrict__ out) {
  __shared__ bf16 aT[128 * 40];
  __shared__ bf16 bT[128 * 40];
  const int tid = threadIdx.x, lane = tid & 31, wv = tid >> 5;
  const int waveM = wv >> 2, waveN = wv & 3;
  const int mT = blockIdx.x, nT = blockIdx.y;      // 64 x 8

  const bf16*  aa = ag + (long long)(mT * 128) * 1024;
  const float* wa = w + (long long)(nT * 128) * 1024;

  v8f acc[4][2] = {};
  for (int kt = 0; kt < 1024; kt += 32) {
    { // A already bf16: straight copy to LDS
      const int row = tid >> 1, col0 = (tid & 1) * 16;
      const bf16* gp = aa + (long long)row * 1024 + kt + col0;
      v8bf* d = (v8bf*)&aT[row * 40 + col0];
      d[0] = ((const v8bf*)gp)[0];
      d[1] = ((const v8bf*)gp)[1];
    }
    stage_f32_bf16(bT, wa + kt, tid);
    if (kt + 32 < 1024) {  // prefetch next K tile
      __builtin_prefetch(aa + (long long)(tid >> 1) * 1024 + kt + 32 + (tid & 1) * 16, 0, 1);
      __builtin_prefetch(wa + (long long)(tid >> 1) * 1024 + kt + 32 + (tid & 1) * 16, 0, 1);
    }
    __syncthreads();
    const v16bf b0 = b_frag(bT, waveN * 32 + (lane & 15), 40, lane);
    const v16bf b1 = b_frag(bT, waveN * 32 + 16 + (lane & 15), 40, lane);
#pragma unroll
    for (int mf = 0; mf < 4; ++mf) {
      const v16bf a = a_frag(&aT[(waveM * 64 + mf * 16 + (lane & 15)) * 40], lane);
      acc[mf][0] = wmma_bf16(a, b0, acc[mf][0]);
      acc[mf][1] = wmma_bf16(a, b1, acc[mf][1]);
    }
    __syncthreads();
  }
#pragma unroll
  for (int mf = 0; mf < 4; ++mf) {
#pragma unroll
    for (int r = 0; r < 8; ++r) {
      const int row = mT * 128 + waveM * 64 + mf * 16 + (lane >> 4) * 8 + r;
      const long long base = (long long)row * 1024 + nT * 128 + waveN * 32;
      out[base + (lane & 15)]      = acc[mf][0][r];
      out[base + 16 + (lane & 15)] = acc[mf][1][r];
    }
  }
}

// ---------------------------------------------------------------------------
// Launch
// ---------------------------------------------------------------------------
extern "C" void kernel_launch(void* const* d_in, const int* in_sizes, int n_in,
                              void* d_out, int out_size, void* d_ws, size_t ws_size,
                              hipStream_t stream) {
  const float* x        = (const float*)d_in[0];
  const float* qkv_w    = (const float*)d_in[1];
  const float* c_proj_w = (const float*)d_in[2];
  float* out = (float*)d_out;

  const long long per = (long long)B_SZ * H_SZ * T_SZ * HD_SZ;  // 8,388,608 elems
  bf16* q = (bf16*)d_ws;
  bf16* k = q + per;
  bf16* v = k + per;
  bf16* o = v + per;   // [b, t, h*128+d], same element count

  qkv_rope_kernel<<<dim3(64, 24), 256, 0, stream>>>(x, qkv_w, q, k, v);
  flash_attn_kernel<<<dim3(32, 16), 256, 0, stream>>>(q, k, v, o);
  proj_kernel<<<dim3(64, 8), 256, 0, stream>>>(o, c_proj_w, out);
}
